// GONET_55903294324882
// MI455X (gfx1250) — matrix operand order repaired and checked by reading, not compiled, not run on hardware
//
#include <hip/hip_runtime.h>
#include <math.h>

typedef __bf16 bf16;
typedef __attribute__((ext_vector_type(16))) __bf16 v16bf;
typedef __attribute__((ext_vector_type(8)))  __bf16 v8bf;
typedef __attribute__((ext_vector_type(8)))  float  v8f;

#define LDS_K 40   // padded row stride in halves (80B rows -> conflict-free b128)

__device__ __forceinline__ v16bf frag_join(v8bf lo, v8bf hi) {
  v16bf r;
#pragma unroll
  for (int i = 0; i < 8; ++i) { r[i] = lo[i]; r[i + 8] = hi[i]; }
  return r;
}

__device__ __forceinline__ int iclamp(int v, int lo, int hi) {
  return v < lo ? lo : (v > hi ? hi : v);
}

// ---------------------------------------------------------------------------
// Implicit-GEMM conv, WMMA bf16->f32. Block = 128 thr (4 waves), tile 64x64,
// each wave computes a 32x32 quadrant = 4 WMMAs / K-step. Weights prepacked
// zero-padded [g][Mgpad][Kpad] so A staging is guard-free b128 traffic.
// KH/KW compile-time -> constant-divisor K-decode. Gather loads use CLAMPED
// coordinates (always-legal address, no predication) + cndmask zeroing.
// ---------------------------------------------------------------------------
template <int KH, int KW>
__global__ __launch_bounds__(128)
void conv_igemm_wmma(const bf16* __restrict__ Wt, const float* __restrict__ bias,
                     const bf16* __restrict__ act, bf16* __restrict__ out,
                     int Mg, int Mgpad, int K, int Kpad,
                     int CinG, int CinTot, int H, int Wd,
                     int stride, int pad,
                     int OH, int OW, int outBatchStride, int chBase, int relu)
{
  constexpr int KHW = KH * KW;
  __shared__ __attribute__((aligned(16))) bf16 As[64][LDS_K];
  __shared__ __attribute__((aligned(16))) bf16 Bs[64][LDS_K];

  const int tid  = threadIdx.x;
  const int lane = tid & 31;
  const int wave = tid >> 5;
  const int wm = (wave >> 1) * 32, wn = (wave & 1) * 32;
  const int half = lane >> 4, lc = lane & 15;
  const int g  = blockIdx.z;
  const int m0 = blockIdx.y * 64;
  const int n0 = blockIdx.x * 64;
  const int HW   = OH * OW;
  const int HWin = H * Wd;

  // staging roles: each thread owns one LDS row and a 16-wide K sub-chunk
  const int srow = tid >> 1;
  const int ksub = (tid & 1) * 16;

  const bf16* Arow = Wt + (size_t)g * Mgpad * Kpad + (size_t)(m0 + srow) * Kpad;

  // implicit-im2col: decode output position once (N is a multiple of 64)
  const int n  = n0 + srow;
  const int bb = n / HW; const int s0 = n - bb * HW;
  const int oy = s0 / OW, ox = s0 - oy * OW;
  const int iy0 = oy * stride - pad;
  const int ix0 = ox * stride - pad;
  const bf16* actB = act + ((size_t)bb * CinTot + g * CinG) * HWin;

  v8f acc00 = {}, acc01 = {}, acc10 = {}, acc11 = {};
  const int ksteps = (K + 31) >> 5;

  for (int ks = 0; ks < ksteps; ++ks) {
    const int k0 = ks << 5;
    // ---- A stage: 2 aligned b128 loads, guard-free (padded weights) ----
    {
      const v8bf* src = (const v8bf*)(Arow + k0 + ksub);
      v8bf a0 = src[0], a1 = src[1];
      __builtin_prefetch(Arow + k0 + 32 + ksub, 0, 3);
      *(v8bf*)&As[srow][ksub]     = a0;
      *(v8bf*)&As[srow][ksub + 8] = a1;
    }
    // ---- B stage: clamped-address gather, incremental (ci,ky,kx) decode ----
    {
      int kk = k0 + ksub;
      int ci = kk / KHW; int r = kk - ci * KHW;
      int ky = r / KW;   int kx = r - ky * KW;
      v8bf t0, t1;
#pragma unroll
      for (int i = 0; i < 8; ++i) {
        int iy = iy0 + ky, ix = ix0 + kx;
        bool ok = (kk + i < K) & ((unsigned)iy < (unsigned)H) & ((unsigned)ix < (unsigned)Wd);
        int iyc = iclamp(iy, 0, H - 1);
        int ixc = iclamp(ix, 0, Wd - 1);
        int cic = ci < CinG - 1 ? ci : CinG - 1;
        bf16 v = actB[cic * HWin + iyc * Wd + ixc];    // always-legal address
        t0[i] = ok ? v : (bf16)0.0f;
        if (++kx == KW) { kx = 0; if (++ky == KH) { ky = 0; ++ci; } }
      }
#pragma unroll
      for (int i = 0; i < 8; ++i) {
        int iy = iy0 + ky, ix = ix0 + kx;
        bool ok = (kk + 8 + i < K) & ((unsigned)iy < (unsigned)H) & ((unsigned)ix < (unsigned)Wd);
        int iyc = iclamp(iy, 0, H - 1);
        int ixc = iclamp(ix, 0, Wd - 1);
        int cic = ci < CinG - 1 ? ci : CinG - 1;
        bf16 v = actB[cic * HWin + iyc * Wd + ixc];
        t1[i] = ok ? v : (bf16)0.0f;
        if (++kx == KW) { kx = 0; if (++ky == KH) { ky = 0; ++ci; } }
      }
      *(v8bf*)&Bs[srow][ksub]     = t0;
      *(v8bf*)&Bs[srow][ksub + 8] = t1;
    }
    __syncthreads();

    // fragments (documented 16-bit WMMA layouts), all ds_load_b128
    v16bf a0 = frag_join(*(const v8bf*)&As[wm + lc][half * 8],
                         *(const v8bf*)&As[wm + lc][16 + half * 8]);
    v16bf a1 = frag_join(*(const v8bf*)&As[wm + 16 + lc][half * 8],
                         *(const v8bf*)&As[wm + 16 + lc][16 + half * 8]);
    v16bf b0 = frag_join(*(const v8bf*)&Bs[wn + lc][half * 8],
                         *(const v8bf*)&Bs[wn + lc][16 + half * 8]);
    v16bf b1 = frag_join(*(const v8bf*)&Bs[wn + 16 + lc][half * 8],
                         *(const v8bf*)&Bs[wn + 16 + lc][16 + half * 8]);

    acc00 = __builtin_amdgcn_wmma_f32_16x16x32_bf16(false, a0, false, b0, (short)0, acc00, false, false);
    acc01 = __builtin_amdgcn_wmma_f32_16x16x32_bf16(false, a0, false, b1, (short)0, acc01, false, false);
    acc10 = __builtin_amdgcn_wmma_f32_16x16x32_bf16(false, a1, false, b0, (short)0, acc10, false, false);
    acc11 = __builtin_amdgcn_wmma_f32_16x16x32_bf16(false, a1, false, b1, (short)0, acc11, false, false);
    __syncthreads();
  }

  // ---- epilogue: bias (+ReLU), NCHW bf16 scatter ----
  v8f accs[2][2] = {{acc00, acc01}, {acc10, acc11}};
#pragma unroll
  for (int j = 0; j < 2; ++j) {
    int col = n0 + wn + j * 16 + lc;                 // always < N
    int b  = col / HW; int s = col - b * HW;
    int oy2 = s / OW;  int ox2 = s - oy2 * OW;
    size_t obase = (size_t)b * outBatchStride + (size_t)oy2 * OW + ox2;
#pragma unroll
    for (int i = 0; i < 2; ++i) {
#pragma unroll
      for (int jj = 0; jj < 8; ++jj) {
        int row = m0 + wm + i * 16 + jj + half * 8;
        if (row < Mg) {
          int cb = g * Mg + row;
          float v = accs[i][j][jj] + bias[cb];
          if (relu) v = v > 0.f ? v : 0.f;
          out[obase + (size_t)(chBase + cb) * HW] = (bf16)v;
        }
      }
    }
  }
}

// ---------------------------------------------------------------------------
// FC GEMM: C[M x N] = A[M x Kpad(padded,zeroed)] * B, with B prepacked
// TRANSPOSED+padded as BT[Npad x Kpad]. Fully guard-free staging.
// ---------------------------------------------------------------------------
template <int F32OUT>
__global__ __launch_bounds__(128)
void fc_gemm_wmma(const bf16* __restrict__ A, int lda,
                  const bf16* __restrict__ BT, int ldb,
                  const float* __restrict__ bias, void* __restrict__ outv,
                  int M, int N, int Kpad, int relu)
{
  __shared__ __attribute__((aligned(16))) bf16 As[64][LDS_K];
  __shared__ __attribute__((aligned(16))) bf16 Bs[64][LDS_K];

  const int tid  = threadIdx.x;
  const int lane = tid & 31;
  const int wave = tid >> 5;
  const int wm = (wave >> 1) * 32, wn = (wave & 1) * 32;
  const int half = lane >> 4, lc = lane & 15;
  const int m0 = blockIdx.y * 64;
  const int n0 = blockIdx.x * 64;

  const int srow = tid >> 1;
  const int ksub = (tid & 1) * 16;
  const bf16* Arow = A  + (size_t)(m0 + srow) * lda;
  const bf16* Brow = BT + (size_t)(n0 + srow) * ldb;

  v8f acc00 = {}, acc01 = {}, acc10 = {}, acc11 = {};
  const int ksteps = Kpad >> 5;

  for (int ks = 0; ks < ksteps; ++ks) {
    const int k0 = ks << 5;
    {
      const v8bf* sa = (const v8bf*)(Arow + k0 + ksub);
      const v8bf* sb = (const v8bf*)(Brow + k0 + ksub);
      v8bf a0 = sa[0], a1 = sa[1];
      v8bf b0 = sb[0], b1 = sb[1];
      __builtin_prefetch(Brow + k0 + 32 + ksub, 0, 3);
      *(v8bf*)&As[srow][ksub]     = a0;
      *(v8bf*)&As[srow][ksub + 8] = a1;
      *(v8bf*)&Bs[srow][ksub]     = b0;
      *(v8bf*)&Bs[srow][ksub + 8] = b1;
    }
    __syncthreads();

    v16bf a0 = frag_join(*(const v8bf*)&As[wm + lc][half * 8],
                         *(const v8bf*)&As[wm + lc][16 + half * 8]);
    v16bf a1 = frag_join(*(const v8bf*)&As[wm + 16 + lc][half * 8],
                         *(const v8bf*)&As[wm + 16 + lc][16 + half * 8]);
    v16bf b0 = frag_join(*(const v8bf*)&Bs[wn + lc][half * 8],
                         *(const v8bf*)&Bs[wn + lc][16 + half * 8]);
    v16bf b1 = frag_join(*(const v8bf*)&Bs[wn + 16 + lc][half * 8],
                         *(const v8bf*)&Bs[wn + 16 + lc][16 + half * 8]);

    acc00 = __builtin_amdgcn_wmma_f32_16x16x32_bf16(false, a0, false, b0, (short)0, acc00, false, false);
    acc01 = __builtin_amdgcn_wmma_f32_16x16x32_bf16(false, a0, false, b1, (short)0, acc01, false, false);
    acc10 = __builtin_amdgcn_wmma_f32_16x16x32_bf16(false, a1, false, b0, (short)0, acc10, false, false);
    acc11 = __builtin_amdgcn_wmma_f32_16x16x32_bf16(false, a1, false, b1, (short)0, acc11, false, false);
    __syncthreads();
  }

  v8f accs[2][2] = {{acc00, acc01}, {acc10, acc11}};
#pragma unroll
  for (int j = 0; j < 2; ++j) {
    int col = n0 + wn + j * 16 + lc;
    if (col < N) {
      float bz = bias[col];
#pragma unroll
      for (int i = 0; i < 2; ++i) {
#pragma unroll
        for (int jj = 0; jj < 8; ++jj) {
          int row = m0 + wm + i * 16 + jj + half * 8;
          if (row < M) {
            float v = accs[i][j][jj] + bz;
            if (relu) v = v > 0.f ? v : 0.f;
            if (F32OUT) ((float*)outv)[(size_t)row * N + col] = v;
            else        ((bf16*)outv)[(size_t)row * N + col] = (bf16)v;
          }
        }
      }
    }
  }
}

// ------------------------- packing / support kernels -----------------------
__global__ void k_f32_to_bf16(const float* __restrict__ s, bf16* __restrict__ d, size_t n) {
  size_t i = (size_t)blockIdx.x * blockDim.x + threadIdx.x;
  if (i < n) d[i] = (bf16)s[i];
}

// row-major f32 [R][K] -> zero-padded bf16 [Rpad][Kpad]
__global__ void k_pack_rows(const float* __restrict__ in, bf16* __restrict__ out,
                            int R, int Rpad, int K, int Kpad) {
  size_t i = (size_t)blockIdx.x * blockDim.x + threadIdx.x;
  size_t total = (size_t)Rpad * Kpad;
  if (i >= total) return;
  int kp = (int)(i % Kpad); int r = (int)(i / Kpad);
  float v = (r < R && kp < K) ? in[(size_t)r * K + kp] : 0.f;
  out[i] = (bf16)v;
}

// f32 [K][N] -> transposed zero-padded bf16 [Npad][Kpad]
__global__ void k_pack_transpose(const float* __restrict__ in, bf16* __restrict__ out,
                                 int K, int N, int Kpad, int Npad) {
  size_t i = (size_t)blockIdx.x * blockDim.x + threadIdx.x;
  size_t total = (size_t)Npad * Kpad;
  if (i >= total) return;
  int kp = (int)(i % Kpad); int n = (int)(i / Kpad);
  float v = (n < N && kp < K) ? in[(size_t)kp * N + n] : 0.f;
  out[i] = (bf16)v;
}

__global__ void k_zero_cols(bf16* __restrict__ X, int rows, int rowStride,
                            int col0, int ncols) {
  int i = blockIdx.x * blockDim.x + threadIdx.x;
  if (i >= rows * ncols) return;
  int r = i / ncols, c = i - (i / ncols) * ncols;
  X[(size_t)r * rowStride + col0 + c] = (bf16)0.0f;
}

__global__ void k_maxpool3s2(const bf16* __restrict__ in, bf16* __restrict__ out,
                             int Bt, int C, int H, int W, int OH, int OW) {
  int i = blockIdx.x * blockDim.x + threadIdx.x;
  int total = Bt * C * OH * OW;
  if (i >= total) return;
  int ox = i % OW; int t = i / OW;
  int oy = t % OH; t /= OH;
  int c = t % C;   int b = t / C;
  const bf16* p = in + (size_t)(b * C + c) * H * W;
  int y0 = oy * 2, x0 = ox * 2;
  float m = -3.4e38f;
#pragma unroll
  for (int dy = 0; dy < 3; ++dy)
#pragma unroll
    for (int dx = 0; dx < 3; ++dx) {
      float v = (float)p[(y0 + dy) * W + (x0 + dx)];
      m = v > m ? v : m;
    }
  out[i] = (bf16)m;
}

__global__ void k_lrn5(const bf16* __restrict__ in, bf16* __restrict__ out,
                       int Bt, int C, int HW) {
  int i = blockIdx.x * blockDim.x + threadIdx.x;
  int total = Bt * C * HW;
  if (i >= total) return;
  int s = i % HW; int t = i / HW;
  int c = t % C;  int b = t / C;
  const bf16* p = in + (size_t)b * C * HW + s;
  float sum = 0.f;
  for (int j = c - 2; j <= c + 2; ++j)
    if (j >= 0 && j < C) { float v = (float)p[(size_t)j * HW]; sum += v * v; }
  float x = (float)p[(size_t)c * HW];
  out[i] = (bf16)(x * powf(1.0f + (1e-4f / 5.0f) * sum, -0.75f));
}

// Correlation (pad 2, disp {-2,0,2}, /C) -> channels 0..8 of X (stride 9568)
__global__ void k_corr(const bf16* __restrict__ ft, const bf16* __restrict__ fi,
                       bf16* __restrict__ X) {
  int i = blockIdx.x * blockDim.x + threadIdx.x;
  if (i >= 64 * 9 * 36) return;
  int x = i % 6; int t = i / 6;
  int y = t % 6; t /= 6;
  int d = t % 9; int b = t / 9;
  int dy = (d / 3) * 2 - 2;
  int dx = (d % 3) * 2 - 2;
  int iy = y + dy, ix = x + dx;
  float s = 0.f;
  if ((unsigned)iy < 6u && (unsigned)ix < 6u) {
    const bf16* pt = ft + (size_t)b * 256 * 36 + y * 6 + x;
    const bf16* pi = fi + (size_t)b * 256 * 36 + iy * 6 + ix;
    for (int c = 0; c < 256; ++c) s += (float)pt[c * 36] * (float)pi[c * 36];
  }
  X[(size_t)b * 9568 + d * 36 + y * 6 + x] = (bf16)(s * (1.0f / 256.0f));
}

// ---------------------------------------------------------------------------
extern "C" void kernel_launch(void* const* d_in, const int* in_sizes, int n_in,
                              void* d_out, int out_size, void* d_ws, size_t ws_size,
                              hipStream_t stream) {
  (void)in_sizes; (void)n_in; (void)out_size; (void)ws_size;
  const float* target = (const float*)d_in[0];
  const float* image  = (const float*)d_in[1];
  const float* c1_w = (const float*)d_in[2];  const float* c1_b = (const float*)d_in[3];
  const float* c2_w = (const float*)d_in[4];  const float* c2_b = (const float*)d_in[5];
  const float* c3_w = (const float*)d_in[6];  const float* c3_b = (const float*)d_in[7];
  const float* c4_w = (const float*)d_in[8];  const float* c4_b = (const float*)d_in[9];
  const float* c5_w = (const float*)d_in[10]; const float* c5_b = (const float*)d_in[11];
  const float* rd_w = (const float*)d_in[12]; const float* rd_b = (const float*)d_in[13];
  const float* f1_w = (const float*)d_in[14]; const float* f1_b = (const float*)d_in[15];
  const float* f2_w = (const float*)d_in[16]; const float* f2_b = (const float*)d_in[17];
  const float* f3_w = (const float*)d_in[18]; const float* f3_b = (const float*)d_in[19];
  const float* f4_w = (const float*)d_in[20]; const float* f4_b = (const float*)d_in[21];

  char* ws = (char*)d_ws;
  size_t off = 0;
  auto balloc = [&](size_t elems) -> bf16* {
    bf16* p = (bf16*)(ws + off);
    off += ((elems * sizeof(bf16) + 255) & ~(size_t)255);
    return p;
  };

  const size_t IMG = (size_t)64 * 3 * 227 * 227;
  bf16* act0 = balloc(2 * IMG);                        // ping (also feat buffer)
  bf16* act1 = balloc((size_t)128 * 96 * 55 * 55);     // pong (largest conv out)
  bf16* w1 = balloc((size_t)128 * 384);                // Mgpad=128, Kpad=384
  bf16* w2 = balloc((size_t)256 * 1200);
  bf16* w3 = balloc((size_t)384 * 2304);
  bf16* w4 = balloc((size_t)384 * 1728);
  bf16* w5 = balloc((size_t)256 * 1728);
  bf16* wr = balloc((size_t)256 * 256);
  bf16* wf1 = balloc((size_t)4096 * 9568);             // transposed [N][Kpad]
  bf16* wf2 = balloc((size_t)4096 * 4096);
  bf16* wf3 = balloc((size_t)4096 * 4096);
  bf16* wf4 = balloc((size_t)64 * 4096);               // Npad=64
  bf16* X   = balloc((size_t)64 * 9568);               // fc1 input, padded K
  bf16* fcA = balloc((size_t)64 * 4096);
  bf16* fcB = balloc((size_t)64 * 4096);

  auto cvt = [&](const float* s, bf16* d, size_t n) {
    k_f32_to_bf16<<<(unsigned)((n + 255) / 256), 256, 0, stream>>>(s, d, n);
  };
  auto pack = [&](const float* s, bf16* d, int R, int Rpad, int K, int Kpad) {
    size_t n = (size_t)Rpad * Kpad;
    k_pack_rows<<<(unsigned)((n + 255) / 256), 256, 0, stream>>>(s, d, R, Rpad, K, Kpad);
  };
  auto packT = [&](const float* s, bf16* d, int K, int N, int Kpad, int Npad) {
    size_t n = (size_t)Npad * Kpad;
    k_pack_transpose<<<(unsigned)((n + 255) / 256), 256, 0, stream>>>(s, d, K, N, Kpad, Npad);
  };

  cvt(target, act0, IMG);
  cvt(image,  act0 + IMG, IMG);
  pack(c1_w, w1, 96, 128, 363, 384);
  pack(c2_w, w2, 256, 256, 1200, 1200);
  pack(c3_w, w3, 384, 384, 2304, 2304);
  pack(c4_w, w4, 384, 384, 1728, 1728);
  pack(c5_w, w5, 256, 256, 1728, 1728);
  pack(rd_w, wr, 256, 256, 256, 256);
  packT(f1_w, wf1, 9540, 4096, 9568, 4096);
  packT(f2_w, wf2, 4096, 4096, 4096, 4096);
  packT(f3_w, wf3, 4096, 4096, 4096, 4096);
  packT(f4_w, wf4, 4096, 4, 4096, 64);
  k_zero_cols<<<(64 * 28 + 255) / 256, 256, 0, stream>>>(X, 64, 9568, 9540, 28);

  dim3 blk(128);
  auto conv = [&](auto kdim, const bf16* W, const float* b, const bf16* in, bf16* out,
                  int Mg, int Mgpad, int groups, int CinG, int CinTot,
                  int H, int Wd, int st, int pad, int OH, int OW,
                  int outBatchStride, int chBase, int relu, int Nbatch) {
    constexpr int KD = decltype(kdim)::value;
    int N = Nbatch * OH * OW;
    int K = CinG * KD * KD;
    int Kpad = (K + 31) & ~31;
    dim3 grd(N / 64, Mgpad / 64, groups);
    conv_igemm_wmma<KD, KD><<<grd, blk, 0, stream>>>(W, b, in, out, Mg, Mgpad, K, Kpad,
                                                     CinG, CinTot, H, Wd, st, pad,
                                                     OH, OW, outBatchStride, chBase, relu);
  };
  using k11 = std::integral_constant<int, 11>;
  using k5  = std::integral_constant<int, 5>;
  using k3  = std::integral_constant<int, 3>;
  using k1  = std::integral_constant<int, 1>;

  // --- AlexNet on fused batch of 128 (target tower ++ image tower) ---
  conv(k11{}, w1, c1_b, act0, act1, 96, 128, 1, 3, 3, 227, 227, 4, 0, 55, 55, 96 * 3025, 0, 1, 128);
  { int t = 128 * 96 * 27 * 27;
    k_maxpool3s2<<<(t + 255) / 256, 256, 0, stream>>>(act1, act0, 128, 96, 55, 55, 27, 27);
    k_lrn5<<<(t + 255) / 256, 256, 0, stream>>>(act0, act1, 128, 96, 27 * 27); }
  conv(k5{}, w2, c2_b, act1, act0, 128, 128, 2, 48, 96, 27, 27, 1, 2, 27, 27, 256 * 729, 0, 1, 128);
  { int t = 128 * 256 * 13 * 13;
    k_maxpool3s2<<<(t + 255) / 256, 256, 0, stream>>>(act0, act1, 128, 256, 27, 27, 13, 13);
    k_lrn5<<<(t + 255) / 256, 256, 0, stream>>>(act1, act0, 128, 256, 13 * 13); }
  conv(k3{}, w3, c3_b, act0, act1, 384, 384, 1, 256, 256, 13, 13, 1, 1, 13, 13, 384 * 169, 0, 1, 128);
  conv(k3{}, w4, c4_b, act1, act0, 192, 192, 2, 192, 384, 13, 13, 1, 1, 13, 13, 384 * 169, 0, 1, 128);
  conv(k3{}, w5, c5_b, act0, act1, 128, 128, 2, 192, 384, 13, 13, 1, 1, 13, 13, 256 * 169, 0, 1, 128);
  { int t = 128 * 256 * 6 * 6;
    k_maxpool3s2<<<(t + 255) / 256, 256, 0, stream>>>(act1, act0, 128, 256, 13, 13, 6, 6); }

  // --- correlation (ft = batches 0..63, fi = 64..127) + redir -> X ---
  k_corr<<<(64 * 9 * 36 + 255) / 256, 256, 0, stream>>>(act0, act0 + (size_t)64 * 256 * 36, X);
  conv(k1{}, wr, rd_b, act0, X, 256, 256, 1, 256, 256, 6, 6, 1, 0, 6, 6, 9568, 9, 0, 64);

  // --- regressor ---
  dim3 g1(64, 1), g4(1, 1);
  fc_gemm_wmma<0><<<g1, blk, 0, stream>>>(X,   9568, wf1, 9568, f1_b, fcA,   64, 4096, 9568, 1);
  fc_gemm_wmma<0><<<g1, blk, 0, stream>>>(fcA, 4096, wf2, 4096, f2_b, fcB,   64, 4096, 4096, 1);
  fc_gemm_wmma<0><<<g1, blk, 0, stream>>>(fcB, 4096, wf3, 4096, f3_b, fcA,   64, 4096, 4096, 1);
  fc_gemm_wmma<1><<<g4, blk, 0, stream>>>(fcA, 4096, wf4, 4096, f4_b, d_out, 64, 4,    4096, 0);
}
